// AGCGRU_11630771437858
// MI455X (gfx1250) — matrix operand order repaired and checked by reading, not compiled, not run on hardware
//
#include <hip/hip_runtime.h>
#include <hip/hip_bf16.h>
#include <math.h>

// ---------------- problem constants ----------------
#define NTOT  2048   // nodes
#define B_    16     // batch
#define L_    12     // sequence length
#define H_    64     // hidden
#define TWOH  128
#define CAT_  66     // IN_DIM + H_DIM
#define CATP  80     // cat padded to 5*16 for WMMA N-tiles
#define KC    198    // CHEB_K * CAT
#define KCP   224    // padded contraction dim (7 * 32)
#define EMB_  10

typedef __attribute__((ext_vector_type(16))) __bf16 v16bf;
typedef __attribute__((ext_vector_type(8)))  float  v8f;

union Frag { uint4 u[2]; v16bf v; };

static __device__ __forceinline__ v8f wmma_bf16(v16bf a, v16bf b, v8f c) {
  return __builtin_amdgcn_wmma_f32_16x16x32_bf16(false, a, false, b, (short)0, c,
                                                 false, false);
}

// CDNA5 async DMA: global -> LDS, tracked by ASYNCcnt (no VGPR round trip).
static __device__ __forceinline__ void async_load_b128(unsigned lds_off, const void* g) {
  unsigned long long ga = (unsigned long long)g;
  asm volatile("global_load_async_to_lds_b128 %0, %1, off"
               :: "v"(lds_off), "v"(ga) : "memory");
}
static __device__ __forceinline__ void wait_async() {
  asm volatile("s_wait_asynccnt 0" ::: "memory");
}
// low 32 bits of a flat shared-memory address == LDS byte offset (flat aperture rule)
static __device__ __forceinline__ unsigned lds_off32(const void* p) {
  return (unsigned)(size_t)p;
}

// ---------------- one-time prep kernels ----------------

__global__ void zero_h_kernel(float* __restrict__ h) {
  size_t idx = (size_t)blockIdx.x * blockDim.x + threadIdx.x;
  if (idx < (size_t)B_ * NTOT * H_) h[idx] = 0.0f;
}

// bf16 copy of A plus a TRANSPOSED bf16 copy (so the T2 GEMM's B operand is
// K-contiguous per column -> b128 LDS fragments instead of u16 gathers)
__global__ void cvt_A_kernel(const float* __restrict__ A, __bf16* __restrict__ Abf,
                             __bf16* __restrict__ At) {
  size_t idx = (size_t)blockIdx.x * blockDim.x + threadIdx.x;
  if (idx < (size_t)NTOT * NTOT) {
    int r = (int)(idx / NTOT), c = (int)(idx % NTOT);
    __bf16 v = (__bf16)A[idx];
    Abf[idx] = v;
    At[(size_t)c * NTOT + r] = v;
  }
}

// T2 = 2*A@A - I  (one-time; double-buffered async-to-LDS, all-b128 fragments)
__global__ __launch_bounds__(128) void gemm_T2_kernel(const __bf16* __restrict__ Abf,
                                                      const __bf16* __restrict__ At,
                                                      __bf16* __restrict__ T2bf) {
  __shared__ __bf16 aT[2][64][32];   // rows x K
  __shared__ __bf16 bT[2][64][32];   // cols x K (from transposed copy)
  const int tid  = threadIdx.x;
  const int lane = tid & 31;
  const int wave = tid >> 5;
  const int row0 = blockIdx.x * 64;
  const int col0 = blockIdx.y * 64;

  auto stage = [&](int sel, int kk) {
    int r = tid >> 1, c = (tid & 1) * 16;
    {
      const __bf16* g = Abf + (size_t)(row0 + r) * NTOT + kk + c;
      unsigned d = lds_off32(&aT[sel][r][c]);
      async_load_b128(d, g);
      async_load_b128(d + 16, g + 8);
    }
    {
      const __bf16* g = At + (size_t)(col0 + r) * NTOT + kk + c;
      unsigned d = lds_off32(&bT[sel][r][c]);
      async_load_b128(d, g);
      async_load_b128(d + 16, g + 8);
    }
  };

  v8f acc[4] = {};
  stage(0, 0);
  wait_async();
  __syncthreads();
  int sel = 0;
  for (int kk = 0; kk < NTOT; kk += 32) {
    if (kk + 32 < NTOT) stage(sel ^ 1, kk + 32);
    Frag af;
    {
      int m = lane & 15, kb = (lane < 16) ? 0 : 8;
      const __bf16* base = &aT[sel][wave * 16 + m][0];
      af.u[0] = *(const uint4*)(base + kb);
      af.u[1] = *(const uint4*)(base + kb + 16);
    }
#pragma unroll
    for (int nt = 0; nt < 4; ++nt) {
      Frag bfr;
      int col = nt * 16 + (lane & 15), kb2 = (lane < 16) ? 0 : 8;
      const __bf16* base = &bT[sel][col][kb2];
      bfr.u[0] = *(const uint4*)(base);
      bfr.u[1] = *(const uint4*)(base + 16);
      acc[nt] = wmma_bf16(af.v, bfr.v, acc[nt]);
    }
    wait_async();
    __syncthreads();
    sel ^= 1;
  }
#pragma unroll
  for (int nt = 0; nt < 4; ++nt) {
    int ncol  = col0 + nt * 16 + (lane & 15);
    int rbase = row0 + wave * 16 + ((lane < 16) ? 0 : 8);
#pragma unroll
    for (int r = 0; r < 8; ++r) {
      int nrow = rbase + r;
      float v = 2.0f * acc[nt][r] - ((nrow == ncol) ? 1.0f : 0.0f);
      T2bf[(size_t)nrow * NTOT + ncol] = (__bf16)v;
    }
  }
}

// node-adaptive weights, TRANSPOSED for fragment-friendly loads:
// Wt[n, o, c] = sum_d emb[n,d] * pool[(d*198 + c)*Co + o]   (c < 198, else 0)
__global__ void mat_W_kernel(const float* __restrict__ emb, const float* __restrict__ pool,
                             __bf16* __restrict__ Wt, int Co) {
  size_t total = (size_t)NTOT * Co * KCP;
  for (size_t idx = (size_t)blockIdx.x * blockDim.x + threadIdx.x; idx < total;
       idx += (size_t)gridDim.x * blockDim.x) {
    int c = (int)(idx % KCP);
    size_t rem = idx / KCP;
    int o = (int)(rem % Co);
    int n = (int)(rem / Co);
    float acc = 0.0f;
    if (c < KC) {
      const float* pp = pool + (size_t)c * Co + o;
      const float* ep = emb + n * EMB_;
#pragma unroll
      for (int d = 0; d < EMB_; ++d) acc += ep[d] * pp[(size_t)d * KC * Co];
    }
    Wt[idx] = (__bf16)acc;
  }
}

__global__ void mat_b_kernel(const float* __restrict__ emb, const float* __restrict__ pool,
                             float* __restrict__ bias, int Co) {
  int idx = blockIdx.x * blockDim.x + threadIdx.x;
  if (idx >= NTOT * Co) return;
  int o = idx % Co, n = idx / Co;
  float acc = 0.0f;
#pragma unroll
  for (int d = 0; d < EMB_; ++d) acc += emb[n * EMB_ + d] * pool[d * Co + o];
  bias[idx] = acc;
}

// ---------------- per-step kernels ----------------

// cat stored CHANNEL-MAJOR: catT[b, c, n]; also fills xg k0 slice (row-major) + pads
__global__ void build_cat_kernel(const float* __restrict__ h, const float* __restrict__ x,
                                 __bf16* __restrict__ catT, __bf16* __restrict__ xg, int t) {
  int idx = blockIdx.x * blockDim.x + threadIdx.x;  // b*N + n
  if (idx >= B_ * NTOT) return;
  int bb = idx / NTOT, n = idx % NTOT;
  __bf16* ct = catT + (size_t)bb * CATP * NTOT + n;
  size_t gbase = (size_t)idx * KCP;
  const float* hp = h + (size_t)idx * H_;
  for (int c = 0; c < H_; ++c) {
    __bf16 v = (__bf16)hp[c];
    ct[(size_t)c * NTOT] = v;
    xg[gbase + c]        = v;
  }
  const float* xp = x + (((size_t)bb * L_ + t) * NTOT + n) * 2;
  __bf16 x0 = (__bf16)xp[0], x1 = (__bf16)xp[1];
  ct[(size_t)64 * NTOT] = x0; ct[(size_t)65 * NTOT] = x1;
  xg[gbase + 64] = x0;        xg[gbase + 65] = x1;
  for (int c = CAT_; c < CATP; ++c) ct[(size_t)c * NTOT] = (__bf16)0.0f;
  for (int c = KC; c < KCP; ++c)    xg[gbase + c]        = (__bf16)0.0f;
}

// xg[b, n, k*66 + j] = (T_k @ cat[b])[n, j]   for k in {1,2}
// 4 waves/block, 64 rows x 80 cols; double-buffered async-to-LDS staging.
__global__ __launch_bounds__(128) void gemm_graph_kernel(
    const __bf16* __restrict__ T1bf, const __bf16* __restrict__ T2bf,
    const __bf16* __restrict__ catT,   // (B, CATP, NTOT) channel-major
    __bf16* __restrict__ xg) {         // (B, NTOT, KCP)
  __shared__ __bf16 aT[2][64][32];     // rows x K
  __shared__ __bf16 bT[2][CATP][32];   // cols x K (transposed: K contiguous per col)
  const int tid  = threadIdx.x;
  const int lane = tid & 31;
  const int wave = tid >> 5;
  const int row0 = blockIdx.x * 64;
  const int k    = blockIdx.y + 1;     // 1 or 2
  const int b    = blockIdx.z;
  const __bf16* T    = (k == 1) ? T1bf : T2bf;
  const __bf16* catB = catT + (size_t)b * CATP * NTOT;

  auto stage = [&](int sel, int kk) {
    // T slab 64x32: 2 x b128 per thread
    int r = tid >> 1, c = (tid & 1) * 16;
    const __bf16* gA = T + (size_t)(row0 + r) * NTOT + kk + c;
    unsigned dA = lds_off32(&aT[sel][r][c]);
    async_load_b128(dA, gA);
    async_load_b128(dA + 16, gA + 8);
    // cat slab 80x32 (K contiguous in global thanks to channel-major layout)
    for (int idx = tid; idx < CATP * 4; idx += 128) {
      int rr = idx >> 2, cc = (idx & 3) * 8;
      async_load_b128(lds_off32(&bT[sel][rr][cc]),
                      catB + (size_t)rr * NTOT + kk + cc);
    }
  };

  v8f acc[5] = {};
  stage(0, 0);
  wait_async();
  __syncthreads();
  int sel = 0;
  for (int kk = 0; kk < NTOT; kk += 32) {
    if (kk + 32 < NTOT) stage(sel ^ 1, kk + 32);   // prefetch next slab (ASYNCcnt)
    Frag af;
    {
      int m = lane & 15, kb = (lane < 16) ? 0 : 8;
      const __bf16* base = &aT[sel][wave * 16 + m][0];
      af.u[0] = *(const uint4*)(base + kb);
      af.u[1] = *(const uint4*)(base + kb + 16);
    }
#pragma unroll
    for (int nt = 0; nt < 5; ++nt) {
      Frag bfr;
      int col = nt * 16 + (lane & 15), kb2 = (lane < 16) ? 0 : 8;
      const __bf16* base = &bT[sel][col][kb2];
      bfr.u[0] = *(const uint4*)(base);
      bfr.u[1] = *(const uint4*)(base + 16);
      acc[nt] = wmma_bf16(af.v, bfr.v, acc[nt]);
    }
    wait_async();
    __syncthreads();
    sel ^= 1;
  }
#pragma unroll
  for (int nt = 0; nt < 5; ++nt) {
    int col = nt * 16 + (lane & 15);
    if (col < CAT_) {
      int rbase = row0 + wave * 16 + ((lane < 16) ? 0 : 8);
#pragma unroll
      for (int r = 0; r < 8; ++r) {
        int nrow = rbase + r;
        xg[((size_t)b * NTOT + nrow) * KCP + k * CAT_ + col] = (__bf16)acc[nt][r];
      }
    }
  }
}

// gate = sigmoid(Xg_n (16x224) @ Wg[n] (224x128) + bg[n]); r -> cat2, u -> ubuf.
// One wave per node; all fragments are vectorized b128 global loads.
__global__ __launch_bounds__(256) void gate_kernel(
    const __bf16* __restrict__ xg, const __bf16* __restrict__ Wgt,  // (N, 128, KCP)
    const float* __restrict__ bg, const float* __restrict__ h,
    const float* __restrict__ x, __bf16* __restrict__ catT,
    __bf16* __restrict__ xgk0, float* __restrict__ ubuf, int t) {
  const int lane = threadIdx.x & 31;
  const int wave = threadIdx.x >> 5;
  const int n = blockIdx.x * 8 + wave;
  v8f acc[8] = {};
  const __bf16* Wn = Wgt + (size_t)n * TWOH * KCP;
  for (int kt = 0; kt < KCP; kt += 32) {
    if (kt + 32 < KCP) __builtin_prefetch(Wn + kt + 32, 0, 0);
    Frag af;
    {
      int bb = lane & 15, kb = (lane < 16) ? 0 : 8;
      const __bf16* src = xg + ((size_t)bb * NTOT + n) * KCP + kt + kb;
      af.u[0] = *(const uint4*)(src);
      af.u[1] = *(const uint4*)(src + 16);
    }
#pragma unroll
    for (int nt = 0; nt < 8; ++nt) {
      int o = nt * 16 + (lane & 15), kb2 = (lane < 16) ? 0 : 8;
      const __bf16* wsrc = Wn + (size_t)o * KCP + kt + kb2;
      Frag bfr;
      bfr.u[0] = *(const uint4*)(wsrc);
      bfr.u[1] = *(const uint4*)(wsrc + 16);
      acc[nt] = wmma_bf16(af.v, bfr.v, acc[nt]);
    }
  }
  const int c16 = lane & 15;
  const int boff = (lane < 16) ? 0 : 8;
#pragma unroll
  for (int nt = 0; nt < 8; ++nt) {
    int o = nt * 16 + c16;
    float bgv = bg[n * TWOH + o];
#pragma unroll
    for (int r = 0; r < 8; ++r) {
      int bb = r + boff;
      float g = 1.0f / (1.0f + expf(-(acc[nt][r] + bgv)));
      if (o < H_) {  // r-gate: cat2 = [r*h, x]
        float hv = h[((size_t)bb * NTOT + n) * H_ + o];
        __bf16 pv = (__bf16)(g * hv);
        catT[((size_t)bb * CATP + o) * NTOT + n] = pv;
        xgk0[((size_t)bb * NTOT + n) * KCP + o]  = pv;
      } else {       // u-gate
        ubuf[((size_t)bb * NTOT + n) * H_ + (o - H_)] = g;
      }
    }
  }
  if (lane < 16) {   // x channels + pads for cat2 / xg(k0)
    int bb = lane;
    const float* xp = x + (((size_t)bb * L_ + t) * NTOT + n) * 2;
    __bf16 x0 = (__bf16)xp[0], x1 = (__bf16)xp[1];
    __bf16* ct = catT + (size_t)bb * CATP * NTOT + n;
    size_t gbase = ((size_t)bb * NTOT + n) * KCP;
    ct[(size_t)64 * NTOT] = x0; ct[(size_t)65 * NTOT] = x1;
    xgk0[gbase + 64] = x0;      xgk0[gbase + 65] = x1;
    for (int c = CAT_; c < CATP; ++c) ct[(size_t)c * NTOT] = (__bf16)0.0f;
    for (int c = KC; c < KCP; ++c)    xgk0[gbase + c]      = (__bf16)0.0f;
  }
}

// c = tanh(Xg2_n @ Wc[n] + bc[n]); h = u*h + (1-u)*c; write output[b,t,n,:]
__global__ __launch_bounds__(256) void cand_kernel(
    const __bf16* __restrict__ xg, const __bf16* __restrict__ Wct,  // (N, 64, KCP)
    const float* __restrict__ bc, const float* __restrict__ ubuf,
    float* __restrict__ h, float* __restrict__ out, int t) {
  const int lane = threadIdx.x & 31;
  const int wave = threadIdx.x >> 5;
  const int n = blockIdx.x * 8 + wave;
  v8f acc[4] = {};
  const __bf16* Wn = Wct + (size_t)n * H_ * KCP;
  for (int kt = 0; kt < KCP; kt += 32) {
    if (kt + 32 < KCP) __builtin_prefetch(Wn + kt + 32, 0, 0);
    Frag af;
    {
      int bb = lane & 15, kb = (lane < 16) ? 0 : 8;
      const __bf16* src = xg + ((size_t)bb * NTOT + n) * KCP + kt + kb;
      af.u[0] = *(const uint4*)(src);
      af.u[1] = *(const uint4*)(src + 16);
    }
#pragma unroll
    for (int nt = 0; nt < 4; ++nt) {
      int o = nt * 16 + (lane & 15), kb2 = (lane < 16) ? 0 : 8;
      const __bf16* wsrc = Wn + (size_t)o * KCP + kt + kb2;
      Frag bfr;
      bfr.u[0] = *(const uint4*)(wsrc);
      bfr.u[1] = *(const uint4*)(wsrc + 16);
      acc[nt] = wmma_bf16(af.v, bfr.v, acc[nt]);
    }
  }
  const int c16 = lane & 15;
  const int boff = (lane < 16) ? 0 : 8;
#pragma unroll
  for (int nt = 0; nt < 4; ++nt) {
    int o = nt * 16 + c16;
    float bcv = bc[n * H_ + o];
#pragma unroll
    for (int r = 0; r < 8; ++r) {
      int bb = r + boff;
      float c = tanhf(acc[nt][r] + bcv);
      size_t hidx = ((size_t)bb * NTOT + n) * H_ + o;
      float u  = ubuf[hidx];
      float hv = h[hidx];
      float hn = u * hv + (1.0f - u) * c;
      h[hidx] = hn;
      out[(((size_t)bb * L_ + t) * NTOT + n) * H_ + o] = hn;
    }
  }
}

// ---------------- host side ----------------

static inline size_t align256(size_t v) { return (v + 255) & ~(size_t)255; }

extern "C" void kernel_launch(void* const* d_in, const int* in_sizes, int n_in,
                              void* d_out, int out_size, void* d_ws, size_t ws_size,
                              hipStream_t stream) {
  (void)in_sizes; (void)n_in; (void)out_size; (void)ws_size;
  const float* x       = (const float*)d_in[0];
  const float* emb     = (const float*)d_in[1];
  const float* A       = (const float*)d_in[2];
  const float* Wg_pool = (const float*)d_in[3];
  const float* bg_pool = (const float*)d_in[4];
  const float* Wc_pool = (const float*)d_in[5];
  const float* bc_pool = (const float*)d_in[6];

  float* out = (float*)d_out;
  float* h   = out + (size_t)B_ * L_ * NTOT * H_;  // h_n lives in the tail of d_out

  char* ws = (char*)d_ws;
  size_t off = 0;
  __bf16* T1bf = (__bf16*)(ws + off); off = align256(off + (size_t)NTOT * NTOT * 2);
  __bf16* Atbf = (__bf16*)(ws + off); off = align256(off + (size_t)NTOT * NTOT * 2);
  __bf16* T2bf = (__bf16*)(ws + off); off = align256(off + (size_t)NTOT * NTOT * 2);
  __bf16* Wgt  = (__bf16*)(ws + off); off = align256(off + (size_t)NTOT * TWOH * KCP * 2);
  __bf16* Wct  = (__bf16*)(ws + off); off = align256(off + (size_t)NTOT * H_ * KCP * 2);
  float*  bgv  = (float*)(ws + off);  off = align256(off + (size_t)NTOT * TWOH * 4);
  float*  bcv  = (float*)(ws + off);  off = align256(off + (size_t)NTOT * H_ * 4);
  __bf16* catT = (__bf16*)(ws + off); off = align256(off + (size_t)B_ * CATP * NTOT * 2);
  __bf16* xg   = (__bf16*)(ws + off); off = align256(off + (size_t)B_ * NTOT * KCP * 2);
  float*  ubuf = (float*)(ws + off);  off = align256(off + (size_t)B_ * NTOT * H_ * 4);

  // --- one-time prep (re-run every call for determinism) ---
  zero_h_kernel<<<(B_ * NTOT * H_ + 255) / 256, 256, 0, stream>>>(h);
  cvt_A_kernel<<<(NTOT * NTOT + 255) / 256, 256, 0, stream>>>(A, T1bf, Atbf);
  gemm_T2_kernel<<<dim3(NTOT / 64, NTOT / 64), 128, 0, stream>>>(T1bf, Atbf, T2bf);
  mat_W_kernel<<<8192, 256, 0, stream>>>(emb, Wg_pool, Wgt, TWOH);
  mat_W_kernel<<<8192, 256, 0, stream>>>(emb, Wc_pool, Wct, H_);
  mat_b_kernel<<<(NTOT * TWOH + 255) / 256, 256, 0, stream>>>(emb, bg_pool, bgv, TWOH);
  mat_b_kernel<<<(NTOT * H_ + 255) / 256, 256, 0, stream>>>(emb, bc_pool, bcv, H_);

  // --- recurrence ---
  for (int t = 0; t < L_; ++t) {
    build_cat_kernel<<<(B_ * NTOT + 255) / 256, 256, 0, stream>>>(h, x, catT, xg, t);
    gemm_graph_kernel<<<dim3(NTOT / 64, 2, B_), 128, 0, stream>>>(T1bf, T2bf, catT, xg);
    gate_kernel<<<NTOT / 8, 256, 0, stream>>>(xg, Wgt, bgv, h, x, catT, xg, ubuf, t);
    gemm_graph_kernel<<<dim3(NTOT / 64, 2, B_), 128, 0, stream>>>(T1bf, T2bf, catT, xg);
    cand_kernel<<<NTOT / 8, 256, 0, stream>>>(xg, Wct, bcv, ubuf, h, out, t);
  }
}
